// LSTM_3874060501274
// MI455X (gfx1250) — matrix operand order repaired and checked by reading, not compiled, not run on hardware
//
#include <hip/hip_runtime.h>
#include <hip/hip_bf16.h>
#include <cstdint>
#include <cstddef>

#define B_  128
#define S_  512
#define I_  512
#define H_  1024
#define G4  4096   // 4*H
#define KC  1536   // I + H  (fused [x_t | h_prev] reduction)

typedef __attribute__((ext_vector_type(16))) __bf16 v16bf;
typedef __attribute__((ext_vector_type(8)))  float  v8f;
typedef __attribute__((ext_vector_type(4)))  unsigned int v4u;
typedef __attribute__((ext_vector_type(8)))  int v8i_;
typedef __attribute__((ext_vector_type(4)))  int v4i_;

// ---------------- helpers ----------------

__device__ __forceinline__ uint16_t f2bf_bits(float f) {
  uint32_t u = __builtin_bit_cast(uint32_t, f);
  u += 0x7fffu + ((u >> 16) & 1u);      // round-to-nearest-even
  return (uint16_t)(u >> 16);
}

__device__ __forceinline__ __bf16 f2bf(float f) {
  uint16_t h = f2bf_bits(f);
  return __builtin_bit_cast(__bf16, h);
}

__device__ __forceinline__ float sigm(float x) {
  return 1.0f / (1.0f + __expf(-x));
}

union FragA { v16bf v; uint4 q[2]; };

#define WMMA_BF16(A, Bv, C) \
  __builtin_amdgcn_wmma_f32_16x16x32_bf16(false, (A), false, (Bv), (short)0, (C), false, false)

// Load 4 B-tiles (16 bf16 each lane, 2 x b128) at K-offset kb.
__device__ __forceinline__ void load_b4(FragA& f0, FragA& f1, FragA& f2, FragA& f3,
                                        const __bf16* __restrict__ base, int kb,
                                        size_t TS) {
  f0.q[0] = *(const uint4*)(base + kb);          f0.q[1] = *(const uint4*)(base + kb + 8);
  f1.q[0] = *(const uint4*)(base + TS + kb);     f1.q[1] = *(const uint4*)(base + TS + kb + 8);
  f2.q[0] = *(const uint4*)(base + 2 * TS + kb); f2.q[1] = *(const uint4*)(base + 2 * TS + kb + 8);
  f3.q[0] = *(const uint4*)(base + 3 * TS + kb); f3.q[1] = *(const uint4*)(base + 3 * TS + kb + 8);
}

#if __has_builtin(__builtin_amdgcn_tensor_load_to_lds)
#define HAVE_TDM 1
// 2-D TDM load: `rows` rows of `dim0` bf16 elements (row stride `stride_elems`)
// from global `gptr` into LDS at byte offset `lds_addr`, packed contiguously.
__device__ __forceinline__ void tdm_load_2d(uint32_t lds_addr, const void* gptr,
                                            uint32_t dim0, uint32_t rows,
                                            uint32_t stride_elems) {
  uint64_t ga = (uint64_t)(uintptr_t)gptr;
  v4u g0 = { 1u,                                   // count=1, user descriptor
             lds_addr,                             // LDS byte address
             (uint32_t)ga,                         // global_addr[31:0]
             (uint32_t)((ga >> 32) & 0x01FFFFFFu) | (2u << 30) }; // addr[56:32] | type=2
  v8i_ g1 = { (int)(1u << 16),                     // wg_mask=0, data_size=1 (2 bytes)
              (int)(dim0 << 16),                   // tensor_dim0[15:0]
              (int)((dim0 >> 16) | (rows << 16)),  // tensor_dim0[31:16] | tensor_dim1[15:0]
              (int)(dim0 << 16),                   // tensor_dim1[31:16]=0 | tile_dim0
              (int)rows,                           // tile_dim1 | tile_dim2=0
              (int)stride_elems,                   // tensor_dim0_stride[31:0]
              0, 0 };
  v4i_ z4 = { 0, 0, 0, 0 };
#if defined(__clang_major__) && (__clang_major__ >= 23)
  v8i_ z8 = { 0, 0, 0, 0, 0, 0, 0, 0 };
  __builtin_amdgcn_tensor_load_to_lds(g0, g1, z4, z4, z8, 0);
#else
  __builtin_amdgcn_tensor_load_to_lds(g0, g1, z4, z4, 0);
#endif
}
#else
#define HAVE_TDM 0
#endif

// ---------------- prep kernels ----------------

__global__ void lstm_pack_WU(const float* __restrict__ Wf, const float* __restrict__ Wi,
                             const float* __restrict__ Wc, const float* __restrict__ Wo,
                             const float* __restrict__ Uf, const float* __restrict__ Ui,
                             const float* __restrict__ Uc, const float* __restrict__ Uo,
                             __bf16* __restrict__ Wz) {
  int idx = blockIdx.x * blockDim.x + threadIdx.x;
  if (idx >= G4 * KC) return;
  int g = idx / KC, k = idx - g * KC;
  int gate = g >> 10, r = g & 1023;
  const float* Wp = (gate == 0) ? Wf : (gate == 1) ? Wi : (gate == 2) ? Wc : Wo;
  const float* Up = (gate == 0) ? Uf : (gate == 1) ? Ui : (gate == 2) ? Uc : Uo;
  float v = (k < I_) ? Wp[r * I_ + k] : Up[r * H_ + (k - I_)];
  Wz[idx] = f2bf(v);
}

__global__ void lstm_pack_pred(const float* __restrict__ pw, __bf16* __restrict__ Pz) {
  int idx = blockIdx.x * blockDim.x + threadIdx.x;
  if (idx >= I_ * H_) return;
  Pz[idx] = f2bf(pw[idx]);
}

__global__ void lstm_pack_x(const float* __restrict__ x, __bf16* __restrict__ xb) {
  size_t i = (size_t)blockIdx.x * blockDim.x + threadIdx.x;
  size_t n4 = (size_t)B_ * S_ * I_ / 4;
  if (i >= n4) return;
  float4 v = ((const float4*)x)[i];
  ushort4 o;
  o.x = f2bf_bits(v.x); o.y = f2bf_bits(v.y);
  o.z = f2bf_bits(v.z); o.w = f2bf_bits(v.w);
  ((ushort4*)xb)[i] = o;
}

__global__ void lstm_zero_state(__bf16* __restrict__ hz, float* __restrict__ c) {
  int i = blockIdx.x * blockDim.x + threadIdx.x;
  if (i >= B_ * H_) return;
  hz[i] = __bf16(0.0f);
  c[i]  = 0.0f;
}

// ---------------- recurrent step ----------------
// grid (8, 16): blockIdx.x = b-tile (16 rows), blockIdx.y = 64-wide h-slab.
// 4 waves; wave = gate. A panel staged to LDS by TDM; each wave computes
// 4 x (16x16) output tiles with double-buffered global B fragments.
__global__ __launch_bounds__(128)
void lstm_step(const __bf16* __restrict__ Wz,     // [4096][1536]
               const __bf16* __restrict__ xbf,    // [B][S][I]
               const __bf16* __restrict__ hprev,  // [B][H]
               __bf16* __restrict__ hnew,         // [B][H]
               float*  __restrict__ c,            // [B][H], updated in place
               const float* __restrict__ bfg, const float* __restrict__ big,
               const float* __restrict__ bcg, const float* __restrict__ bog,
               float* __restrict__ i_out,         // i_hist[t]: [B][H]
               float* __restrict__ f_out,         // f_hist[t]: [B][H]
               int t) {
  __shared__ __bf16 sAx[16 * I_];        // 16 KB: x_t panel, row-major [16][512]
  __shared__ __bf16 sAh[16 * H_];        // 32 KB: h_prev panel, row-major [16][1024]
  __shared__ float  zbuf[4][16 * 64];    // 16 KB: gate exchange

  const int lane = threadIdx.x & 31;
  const int wave = threadIdx.x >> 5;   // gate index 0..3 (f,i,C,o)
  const int half = lane >> 4;
  const int mr   = lane & 15;
  const int b0   = blockIdx.x * 16;
  const int h0   = blockIdx.y * 64;

  const __bf16* xg = xbf + (size_t)b0 * (S_ * I_) + (size_t)t * I_;
  const __bf16* hg = hprev + (size_t)b0 * H_;

#if HAVE_TDM
  if (wave == 0) {
    tdm_load_2d((uint32_t)(uintptr_t)(void*)sAx, xg, I_, 16, S_ * I_);
    tdm_load_2d((uint32_t)(uintptr_t)(void*)sAh, hg, H_, 16, H_);
    __builtin_amdgcn_s_wait_tensorcnt(0);
  }
#else
  for (int i = threadIdx.x * 8; i < 16 * I_; i += 128 * 8) {
    int r = i >> 9, cc = i & (I_ - 1);
    *(uint4*)&sAx[i] = *(const uint4*)(xg + (size_t)r * (S_ * I_) + cc);
  }
  for (int i = threadIdx.x * 8; i < 16 * H_; i += 128 * 8) {
    int r = i >> 10, cc = i & (H_ - 1);
    *(uint4*)&sAh[i] = *(const uint4*)(hg + (size_t)r * H_ + cc);
  }
#endif
  __syncthreads();

  const size_t TS = (size_t)16 * KC;   // B tile stride (g-rows) in elements
  const __bf16* bbase = Wz + (size_t)(wave * H_ + h0 + mr) * KC + half * 16;

  v8f acc0 = {}, acc1 = {}, acc2 = {}, acc3 = {};
  FragA cw0, cw1, cw2, cw3, nw0, nw1, nw2, nw3;

  // ---- x phase: K in [0, 512) ----
  const __bf16* ax = sAx + mr * I_ + half * 16;
  load_b4(cw0, cw1, cw2, cw3, bbase, 0, TS);
#pragma unroll 2
  for (int kb = 0; kb < I_; kb += 32) {
    // prefetch next K-slab (kb=480 prefetch lands exactly on h-phase K=0 weights)
    load_b4(nw0, nw1, nw2, nw3, bbase, kb + 32, TS);
    FragA a;
    a.q[0] = *(const uint4*)(ax + kb);
    a.q[1] = *(const uint4*)(ax + kb + 8);
    acc0 = WMMA_BF16(a.v, cw0.v, acc0);
    acc1 = WMMA_BF16(a.v, cw1.v, acc1);
    acc2 = WMMA_BF16(a.v, cw2.v, acc2);
    acc3 = WMMA_BF16(a.v, cw3.v, acc3);
    cw0 = nw0; cw1 = nw1; cw2 = nw2; cw3 = nw3;
  }

  // ---- h phase: K in [512, 1536) ----
  const __bf16* ah = sAh + mr * H_ + half * 16;
  const __bf16* b2 = bbase + I_;
#pragma unroll 2
  for (int kb = 0; kb < H_; kb += 32) {
    load_b4(nw0, nw1, nw2, nw3, b2, kb + 32, TS);  // tail prefetch stays in ws
    FragA a;
    a.q[0] = *(const uint4*)(ah + kb);
    a.q[1] = *(const uint4*)(ah + kb + 8);
    acc0 = WMMA_BF16(a.v, cw0.v, acc0);
    acc1 = WMMA_BF16(a.v, cw1.v, acc1);
    acc2 = WMMA_BF16(a.v, cw2.v, acc2);
    acc3 = WMMA_BF16(a.v, cw3.v, acc3);
    cw0 = nw0; cw1 = nw1; cw2 = nw2; cw3 = nw3;
  }

  // D layout: reg r, lanes 0-15 -> (M=r, N=lane); lanes 16-31 -> (M=8+r, N=lane-16)
#pragma unroll
  for (int r = 0; r < 8; ++r) {
    int m = r + half * 8;
    zbuf[wave][m * 64 +  0 + mr] = acc0[r];
    zbuf[wave][m * 64 + 16 + mr] = acc1[r];
    zbuf[wave][m * 64 + 32 + mr] = acc2[r];
    zbuf[wave][m * 64 + 48 + mr] = acc3[r];
  }
  __syncthreads();

  // fused gate epilogue: 1024 slab elements / 128 threads
  for (int e = threadIdx.x; e < 16 * 64; e += 128) {
    int m = e >> 6, col = e & 63;
    int bb = b0 + m, hh = h0 + col;
    size_t off = (size_t)bb * H_ + hh;
    float zf = zbuf[0][e] + bfg[hh];
    float zi = zbuf[1][e] + big[hh];
    float zc = zbuf[2][e] + bcg[hh];
    float zo = zbuf[3][e] + bog[hh];
    float ft = sigm(zf);
    float it = sigm(zi);
    float cn = ft * c[off] + it * tanhf(zc);
    float hn = sigm(zo) * tanhf(cn);
    c[off]    = cn;
    hnew[off] = f2bf(hn);
    i_out[off] = it;
    f_out[off] = ft;
  }
}

// ---------------- predictor GEMM ----------------
// out[b][s][n] = h_hist[s*B+b, :] . pred_w[n, :] + pred_b[n]
// grid (S*B/16, 2); 4 waves, each wave 4 N-tiles (64 cols), double-buffered.
__global__ __launch_bounds__(128)
void lstm_pred(const __bf16* __restrict__ hh,   // [S*B][H]
               const __bf16* __restrict__ Pz,   // [I][H]
               const float*  __restrict__ pb,   // [I]
               float* __restrict__ out) {       // [B][S][I]
  const int lane = threadIdx.x & 31;
  const int wave = threadIdx.x >> 5;
  const int half = lane >> 4;
  const int mr   = lane & 15;
  const int m0   = blockIdx.x * 16;
  const int n0   = blockIdx.y * 256 + wave * 64;

  const size_t TS = (size_t)16 * H_;

  const __bf16* arow  = hh + (size_t)(m0 + mr) * H_ + half * 16;
  const __bf16* bbase = Pz + (size_t)(n0 + mr) * H_ + half * 16;

  v8f acc0 = {}, acc1 = {}, acc2 = {}, acc3 = {};
  FragA cw0, cw1, cw2, cw3, nw0, nw1, nw2, nw3;
  load_b4(cw0, cw1, cw2, cw3, bbase, 0, TS);

#pragma unroll 2
  for (int kb = 0; kb < H_; kb += 32) {
    load_b4(nw0, nw1, nw2, nw3, bbase, kb + 32, TS);  // tail prefetch stays in ws
    FragA a;
    a.q[0] = *(const uint4*)(arow + kb);
    a.q[1] = *(const uint4*)(arow + kb + 8);
    acc0 = WMMA_BF16(a.v, cw0.v, acc0);
    acc1 = WMMA_BF16(a.v, cw1.v, acc1);
    acc2 = WMMA_BF16(a.v, cw2.v, acc2);
    acc3 = WMMA_BF16(a.v, cw3.v, acc3);
    cw0 = nw0; cw1 = nw1; cw2 = nw2; cw3 = nw3;
  }

  const int s   = m0 >> 7;
  const int bb0 = m0 & 127;
  float*    orow = out + (size_t)s * I_;
#pragma unroll
  for (int j = 0; j < 4; ++j) {
    const v8f& acc = (j == 0) ? acc0 : (j == 1) ? acc1 : (j == 2) ? acc2 : acc3;
    int n = n0 + j * 16 + mr;
    float bias = pb[n];
#pragma unroll
    for (int r = 0; r < 8; ++r) {
      int bb = bb0 + r + half * 8;
      orow[(size_t)bb * (S_ * I_) + n] = acc[r] + bias;
    }
  }
}

// ---------------- host ----------------

extern "C" void kernel_launch(void* const* d_in, const int* in_sizes, int n_in,
                              void* d_out, int out_size, void* d_ws, size_t ws_size,
                              hipStream_t stream) {
  const float* x   = (const float*)d_in[0];
  const float* Wf  = (const float*)d_in[1];
  const float* Wi  = (const float*)d_in[2];
  const float* Wc  = (const float*)d_in[3];
  const float* Wo  = (const float*)d_in[4];
  const float* Uf  = (const float*)d_in[5];
  const float* Ui  = (const float*)d_in[6];
  const float* Uc  = (const float*)d_in[7];
  const float* Uo  = (const float*)d_in[8];
  const float* bf  = (const float*)d_in[9];
  const float* bi  = (const float*)d_in[10];
  const float* bc  = (const float*)d_in[11];
  const float* bo  = (const float*)d_in[12];
  const float* pw  = (const float*)d_in[13];
  const float* pbv = (const float*)d_in[14];

  // workspace layout (bytes)
  constexpr size_t OFF_WZ = 0;
  constexpr size_t SZ_WZ  = (size_t)G4 * KC * 2;        // 12.6 MB
  constexpr size_t OFF_PZ = OFF_WZ + SZ_WZ;
  constexpr size_t SZ_PZ  = (size_t)I_ * H_ * 2;        // 1 MB
  constexpr size_t OFF_XB = OFF_PZ + SZ_PZ;
  constexpr size_t SZ_XB  = (size_t)B_ * S_ * I_ * 2;   // 64 MB
  constexpr size_t OFF_HH = OFF_XB + SZ_XB;
  constexpr size_t SZ_HH  = (size_t)S_ * B_ * H_ * 2;   // 128 MB
  constexpr size_t OFF_HZ = OFF_HH + SZ_HH;
  constexpr size_t SZ_HZ  = (size_t)B_ * H_ * 2;
  constexpr size_t OFF_C  = OFF_HZ + SZ_HZ;

  char* ws = (char*)d_ws;
  __bf16* Wz    = (__bf16*)(ws + OFF_WZ);
  __bf16* Pz    = (__bf16*)(ws + OFF_PZ);
  __bf16* xbf   = (__bf16*)(ws + OFF_XB);
  __bf16* hhist = (__bf16*)(ws + OFF_HH);
  __bf16* hzero = (__bf16*)(ws + OFF_HZ);
  float*  cbuf  = (float*)(ws + OFF_C);

  // --- prep ---
  lstm_pack_WU<<<(G4 * KC + 255) / 256, 256, 0, stream>>>(Wf, Wi, Wc, Wo,
                                                          Uf, Ui, Uc, Uo, Wz);
  lstm_pack_pred<<<(I_ * H_ + 255) / 256, 256, 0, stream>>>(pw, Pz);
  {
    size_t n4 = (size_t)B_ * S_ * I_ / 4;
    lstm_pack_x<<<(unsigned)((n4 + 255) / 256), 256, 0, stream>>>(x, xbf);
  }
  lstm_zero_state<<<(B_ * H_ + 255) / 256, 256, 0, stream>>>(hzero, cbuf);

  // --- recurrent scan ---
  float* out_f   = (float*)d_out;
  float* i_base  = out_f + (size_t)B_ * S_ * I_;
  float* f_base  = i_base + (size_t)S_ * B_ * H_;
  const size_t bh = (size_t)B_ * H_;

  for (int t = 0; t < S_; ++t) {
    const __bf16* hprev = (t == 0) ? hzero : (hhist + (size_t)(t - 1) * bh);
    lstm_step<<<dim3(B_ / 16, H_ / 64), 128, 0, stream>>>(
        Wz, xbf, hprev, hhist + (size_t)t * bh, cbuf,
        bf, bi, bc, bo,
        i_base + (size_t)t * bh, f_base + (size_t)t * bh, t);
  }

  // --- predictor ---
  lstm_pred<<<dim3((S_ * B_) / 16, I_ / 256), 128, 0, stream>>>(hhist, Pz, pbv, out_f);
}
